// MultiHeadAttention_35734127903058
// MI455X (gfx1250) — compile-verified, matrix operand order checked by reading, not run on hardware
//
#include <hip/hip_runtime.h>
#include <hip/hip_bf16.h>

// ---------------------------------------------------------------------------
// MI455X (gfx1250) multi-head attention with RoPE, bf16 WMMA pipeline.
// B=2, T=2048, D_MODEL=768, H=12, D_K=64.
// ---------------------------------------------------------------------------

#define BB 2
#define TT 2048
#define DM 768
#define NH 12
#define DK 64
#define QKVN 2304

typedef __attribute__((ext_vector_type(16))) __bf16 v16bf;
typedef __attribute__((ext_vector_type(8)))  float  v8f;

union Frag16 {
    v16bf v;
    unsigned u[8];
};

__device__ __forceinline__ unsigned short f32_to_bf16(float f) {
    unsigned u = __float_as_uint(f);
    u += 0x7FFFu + ((u >> 16) & 1u);   // round to nearest even
    return (unsigned short)(u >> 16);
}
__device__ __forceinline__ float bf16_to_f32(unsigned short h) {
    return __uint_as_float(((unsigned)h) << 16);
}
__device__ __forceinline__ unsigned pack_bf16x2(float a, float b) {
    return (unsigned)f32_to_bf16(a) | ((unsigned)f32_to_bf16(b) << 16);
}

// K-offset inside a 16-bit A/B operand fragment (16x32), per CDNA5 ISA 7.12.2:
// VGPR v, lanes 0-15 hold K = (v>>2)*16 + (v&3)*2 (+1); lanes 16-31 add 8.
__device__ __forceinline__ int kofs(int lane, int v) {
    return ((v >> 2) << 4) + ((lane & 16) ? 8 : 0) + ((v & 3) << 1);
}

// ---------------------------------------------------------------------------
// Tensor Data Mover: 1-D contiguous tile, global -> LDS. D# per ISA 8.3-8.6.
// This toolchain exposes the 6-arg builtin:
//   (u32x4 g0, i32x8 g1, i32x4 g2, i32x4 g3, i32x8 g4, i32 cpol)
// ---------------------------------------------------------------------------
#define USE_TDM 0
#if defined(__has_builtin)
#if __has_builtin(__builtin_amdgcn_tensor_load_to_lds) && \
    __has_builtin(__builtin_amdgcn_s_wait_tensorcnt)
#undef USE_TDM
#define USE_TDM 1
#endif
#endif

#if USE_TDM
typedef unsigned int u32x4 __attribute__((ext_vector_type(4)));
typedef int          i32x4 __attribute__((ext_vector_type(4)));
typedef int          i32x8 __attribute__((ext_vector_type(8)));

// Copy `ndw` dwords from contiguous global memory to contiguous LDS.
__device__ __forceinline__ void tdm_load_1d(unsigned lds_off, const void* gptr,
                                            unsigned ndw) {
    unsigned long long ga = (unsigned long long)gptr;
    u32x4 g0;
    g0[0] = 1u;                                   // count=1, user descriptor
    g0[1] = lds_off;                              // lds_addr (bytes)
    g0[2] = (unsigned)(ga & 0xFFFFFFFFu);         // global_addr[31:0]
    g0[3] = (unsigned)((ga >> 32) & 0x01FFFFFFu)  // global_addr[56:32]
          | 0x80000000u;                          // type=2 ("image")
    i32x8 g1;
    g1[0] = 0x00020000;                 // data_size=2 (4 bytes), no flags
    g1[1] = (int)((ndw & 0xFFFFu) << 16);         // tensor_dim0[15:0]
    g1[2] = (int)(((ndw >> 16) & 0xFFFFu)         // tensor_dim0[31:16]
          | (1u << 16));                          // tensor_dim1 = 1
    g1[3] = (int)((ndw & 0xFFFFu) << 16);         // tile_dim0 = ndw
    g1[4] = 0;                                    // tile_dim1/2 = 0 (1-D)
    g1[5] = (int)ndw;                             // tensor_dim0_stride[31:0]
    g1[6] = 0;
    g1[7] = 0;
    i32x4 gz4 = {0, 0, 0, 0};
    i32x8 gz8 = {0, 0, 0, 0, 0, 0, 0, 0};
    __builtin_amdgcn_tensor_load_to_lds(g0, g1, gz4, gz4, gz8, 0);
}
#endif

// ---------------------------------------------------------------------------
// Kernel 1: QKV = X @ Wqkv  (M=4096, N=2304, K=768), bf16 WMMA, f32 acc.
// Epilogue scatters bf16 into Q/K/V buffers laid out [B][H][T][DK].
// ---------------------------------------------------------------------------
__global__ __launch_bounds__(128) void qkv_gemm_kernel(
    const float* __restrict__ x, const float* __restrict__ w,
    unsigned short* __restrict__ qb, unsigned short* __restrict__ kb,
    unsigned short* __restrict__ vb)
{
    __shared__ unsigned short As[64][36];   // [m][k] pairs contiguous in k
    __shared__ unsigned short Bst[64][36];  // [n][k] pairs contiguous in k

    const int tid  = threadIdx.x;
    const int wave = tid >> 5;
    const int lane = tid & 31;
    const int wr = wave >> 1, wc = wave & 1;
    const int m0 = blockIdx.y * 64;
    const int n0 = blockIdx.x * 64;

    v8f c[2][2];
    c[0][0] = {}; c[0][1] = {}; c[1][0] = {}; c[1][1] = {};

    for (int kt = 0; kt < DM / 32; ++kt) {
        __syncthreads();
#pragma unroll
        for (int i = 0; i < 4; ++i) {
            int e = tid + i * 128;                  // 512 float4 per tile
            // A tile: 64 rows x 32 k  (8 float4 per row)
            int ar = e >> 3, ak = (e & 7) * 4;
            float4 av = *(const float4*)&x[(size_t)(m0 + ar) * DM + kt * 32 + ak];
            uint2 ap;
            ap.x = pack_bf16x2(av.x, av.y);
            ap.y = pack_bf16x2(av.z, av.w);
            *(uint2*)&As[ar][ak] = ap;
            // B tile: 32 k-rows x 64 n (16 float4 per row), stored transposed
            int br = e >> 4, bc = (e & 15) * 4;
            float4 bv = *(const float4*)&w[(size_t)(kt * 32 + br) * QKVN + n0 + bc];
            Bst[bc + 0][br] = f32_to_bf16(bv.x);
            Bst[bc + 1][br] = f32_to_bf16(bv.y);
            Bst[bc + 2][br] = f32_to_bf16(bv.z);
            Bst[bc + 3][br] = f32_to_bf16(bv.w);
        }
        __syncthreads();

        Frag16 a[2], b[2];
#pragma unroll
        for (int mb = 0; mb < 2; ++mb)
#pragma unroll
            for (int vv = 0; vv < 8; ++vv) {
                int m = wr * 32 + mb * 16 + (lane & 15);
                a[mb].u[vv] = *(const unsigned*)&As[m][kofs(lane, vv)];
            }
#pragma unroll
        for (int nb = 0; nb < 2; ++nb)
#pragma unroll
            for (int vv = 0; vv < 8; ++vv) {
                int n = wc * 32 + nb * 16 + (lane & 15);
                b[nb].u[vv] = *(const unsigned*)&Bst[n][kofs(lane, vv)];
            }
#pragma unroll
        for (int mb = 0; mb < 2; ++mb)
#pragma unroll
            for (int nb = 0; nb < 2; ++nb)
                c[mb][nb] = __builtin_amdgcn_wmma_f32_16x16x32_bf16(
                    false, a[mb].v, false, b[nb].v, (short)0, c[mb][nb],
                    false, false);
    }

    // Epilogue: q/k/v split and head index are uniform per 16-col fragment.
#pragma unroll
    for (int nb = 0; nb < 2; ++nb) {
        int ncol  = n0 + wc * 32 + nb * 16;     // uniform
        int which = ncol / DM;                  // uniform (scalar div)
        int hd    = ncol - which * DM;
        int h     = hd >> 6;                    // uniform head
        int dbase = hd & 63;                    // 0/16/32/48, uniform
        unsigned short* dst = (which == 0) ? qb : (which == 1) ? kb : vb;
        int d = dbase + (lane & 15);
#pragma unroll
        for (int mb = 0; mb < 2; ++mb)
#pragma unroll
            for (int r = 0; r < 8; ++r) {
                int mg = m0 + wr * 32 + mb * 16 + r + ((lane & 16) ? 8 : 0);
                int bidx = mg >> 11, t = mg & 2047;
                dst[((size_t)(bidx * NH + h) * TT + t) * DK + d] =
                    f32_to_bf16(c[mb][nb][r]);
            }
    }
}

// ---------------------------------------------------------------------------
// Kernel 2: in-place RoPE on Q and K (bf16).  dim//4 = 16 nonzero freqs.
// ---------------------------------------------------------------------------
__global__ void rope_kernel(unsigned short* __restrict__ qb,
                            unsigned short* __restrict__ kb)
{
    const int total = 2 * BB * NH * TT * 32;
    int idx = blockIdx.x * blockDim.x + threadIdx.x;
    if (idx >= total) return;
    int j = idx & 31;
    int t = (idx >> 5) & 2047;
    int rest = idx >> 16;                  // tensor*24 + bh  (0..47)
    int tensor = (rest >= BB * NH) ? 1 : 0;
    int bh = rest - tensor * (BB * NH);
    unsigned short* p = tensor ? kb : qb;
    size_t base = ((size_t)bh * TT + t) * DK;

    float c = 1.0f, s = 0.0f;
    if (j < 16) {
        // angular_freq[j] = 10000^(-j/15)
        float af = __expf(-9.210340371976184f * (float)j * (1.0f / 15.0f));
        sincosf(af * (float)t, &s, &c);
    }
    float x1 = bf16_to_f32(p[base + j]);
    float x2 = bf16_to_f32(p[base + 32 + j]);
    p[base + j]      = f32_to_bf16(x1 * c + x2 * s);
    p[base + 32 + j] = f32_to_bf16(-x1 * s + x2 * c);
}

// ---------------------------------------------------------------------------
// Kernel 3: causal flash attention per (b,h).  Block = 8 waves = 128 queries,
// each wave owns 16 queries; K/V tiles of 32 keys staged in LDS.  The K tile
// is a contiguous 4KB span of global memory -> moved by the Tensor Data
// Mover when available.
// ---------------------------------------------------------------------------
__global__ __launch_bounds__(256) void attn_kernel(
    const unsigned short* __restrict__ qbuf,
    const unsigned short* __restrict__ kbuf,
    const unsigned short* __restrict__ vbuf,
    float* __restrict__ ows)
{
    __shared__ unsigned short Ks[32][64];       // [key][d], contiguous (TDM dst)
    __shared__ unsigned short Vt[64][36];       // [d][key] (transposed)
    __shared__ unsigned short Plds[8][16][32];  // per-wave P staging

    const int tid  = threadIdx.x;
    const int wave = tid >> 5;
    const int lane = tid & 31;
    const int qb0  = blockIdx.x * 128;
    const int bh   = blockIdx.y;
    const size_t base = (size_t)bh * TT * DK;
    const int qw0 = qb0 + wave * 16;

    // Q fragments (A-layout, 16 queries x 64 d, as two 16x32 frags), registers.
    Frag16 qf[2];
#pragma unroll
    for (int f = 0; f < 2; ++f)
#pragma unroll
        for (int vv = 0; vv < 8; ++vv) {
            int q = qw0 + (lane & 15);
            int d = kofs(lane, vv) + f * 32;
            qf[f].u[vv] = *(const unsigned*)&qbuf[base + (size_t)q * DK + d];
        }

    v8f acc[4];
    acc[0] = {}; acc[1] = {}; acc[2] = {}; acc[3] = {};
    float rm[8], rl[8];
#pragma unroll
    for (int r = 0; r < 8; ++r) { rm[r] = -3.0e38f; rl[r] = 0.0f; }

    const float SC = 0.125f * 1.44269504088896f;  // 1/sqrt(64) * log2(e)
    const int nkt = (qb0 >> 5) + 4;               // key tiles up to qb0+127

    for (int kt = 0; kt < nkt; ++kt) {
        __syncthreads();
        // Prefetch next tile's K/V cachelines into GL2 (global_prefetch_b8).
        if (kt + 1 < nkt) {
            const unsigned short* nk = kbuf + base + (size_t)(kt + 1) * 32 * DK;
            __builtin_prefetch(nk + (tid << 3), 0, 1);
            const unsigned short* nv = vbuf + base + (size_t)(kt + 1) * 32 * DK;
            __builtin_prefetch(nv + (tid << 3), 0, 1);
        }
#if USE_TDM
        if (wave == 0) {
            tdm_load_1d((unsigned)(unsigned long long)&Ks[0][0],
                        kbuf + base + (size_t)(kt * 32) * DK,
                        (32 * DK * 2) / 4 /* 1024 dwords */);
            __builtin_amdgcn_s_wait_tensorcnt(0);
        }
        {   // V staged (transposed) by all threads: one uint4 each.
            int key = tid >> 3, d8 = (tid & 7) * 8;
            uint4 vv4 = *(const uint4*)&vbuf[base + (size_t)(kt * 32 + key) * DK + d8];
            Vt[d8 + 0][key] = (unsigned short)(vv4.x);
            Vt[d8 + 1][key] = (unsigned short)(vv4.x >> 16);
            Vt[d8 + 2][key] = (unsigned short)(vv4.y);
            Vt[d8 + 3][key] = (unsigned short)(vv4.y >> 16);
            Vt[d8 + 4][key] = (unsigned short)(vv4.z);
            Vt[d8 + 5][key] = (unsigned short)(vv4.z >> 16);
            Vt[d8 + 6][key] = (unsigned short)(vv4.w);
            Vt[d8 + 7][key] = (unsigned short)(vv4.w >> 16);
        }
#else
        {
            int key = tid >> 3, d8 = (tid & 7) * 8;
            size_t g = base + (size_t)(kt * 32 + key) * DK + d8;
            uint4 kk4 = *(const uint4*)&kbuf[g];
            *(uint4*)&Ks[key][d8] = kk4;
            uint4 vv4 = *(const uint4*)&vbuf[g];
            Vt[d8 + 0][key] = (unsigned short)(vv4.x);
            Vt[d8 + 1][key] = (unsigned short)(vv4.x >> 16);
            Vt[d8 + 2][key] = (unsigned short)(vv4.y);
            Vt[d8 + 3][key] = (unsigned short)(vv4.y >> 16);
            Vt[d8 + 4][key] = (unsigned short)(vv4.z);
            Vt[d8 + 5][key] = (unsigned short)(vv4.z >> 16);
            Vt[d8 + 6][key] = (unsigned short)(vv4.w);
            Vt[d8 + 7][key] = (unsigned short)(vv4.w >> 16);
        }
#endif
        __syncthreads();

        // S(16x32) = Q(16x64) @ K^T : two 16x16 C-frags, each 2 WMMAs deep.
        v8f s[2];
        s[0] = {}; s[1] = {};
        Frag16 kf;
#pragma unroll
        for (int j = 0; j < 2; ++j)
#pragma unroll
            for (int f = 0; f < 2; ++f) {
#pragma unroll
                for (int vv = 0; vv < 8; ++vv) {
                    int n = j * 16 + (lane & 15);
                    int kd = kofs(lane, vv) + f * 32;
                    kf.u[vv] = *(const unsigned*)&Ks[n][kd];
                }
                s[j] = __builtin_amdgcn_wmma_f32_16x16x32_bf16(
                    false, qf[f].v, false, kf.v, (short)0, s[j], false, false);
            }

        // Online softmax (row m = r + 8*(lane>=16)); reductions stay inside
        // each 16-lane half, matching the C-matrix row split.
        float p0[8], p1[8], al[8];
#pragma unroll
        for (int r = 0; r < 8; ++r) {
            int m = r + ((lane & 16) ? 8 : 0);
            int qg = qw0 + m;
            int k0g = kt * 32 + (lane & 15);
            float s0 = (k0g      <= qg) ? s[0][r] * SC : -3.0e38f;
            float s1 = (k0g + 16 <= qg) ? s[1][r] * SC : -3.0e38f;
            float mx = fmaxf(s0, s1);
#pragma unroll
            for (int off = 1; off < 16; off <<= 1)
                mx = fmaxf(mx, __shfl_xor(mx, off, 32));
            float mnew = fmaxf(rm[r], mx);
            float alpha = exp2f(rm[r] - mnew);
            rm[r] = mnew;
            float e0 = exp2f(s0 - mnew);
            float e1 = exp2f(s1 - mnew);
            float rs = e0 + e1;
#pragma unroll
            for (int off = 1; off < 16; off <<= 1)
                rs += __shfl_xor(rs, off, 32);
            rl[r] = rl[r] * alpha + rs;
            p0[r] = e0; p1[r] = e1; al[r] = alpha;
        }
#pragma unroll
        for (int db = 0; db < 4; ++db)
#pragma unroll
            for (int r = 0; r < 8; ++r)
                acc[db][r] *= al[r];

        // P: C-layout -> A-layout via per-wave LDS (same-wave DS is in-order).
#pragma unroll
        for (int r = 0; r < 8; ++r) {
            int m = r + ((lane & 16) ? 8 : 0);
            Plds[wave][m][lane & 15]        = f32_to_bf16(p0[r]);
            Plds[wave][m][16 + (lane & 15)] = f32_to_bf16(p1[r]);
        }
        Frag16 pa;
#pragma unroll
        for (int vv = 0; vv < 8; ++vv)
            pa.u[vv] = *(const unsigned*)&Plds[wave][lane & 15][kofs(lane, vv)];

        // acc += P(16x32) @ V(32x64): 4 d-blocks of 16.
        Frag16 vf;
#pragma unroll
        for (int db = 0; db < 4; ++db) {
#pragma unroll
            for (int vv = 0; vv < 8; ++vv)
                vf.u[vv] = *(const unsigned*)&Vt[db * 16 + (lane & 15)][kofs(lane, vv)];
            acc[db] = __builtin_amdgcn_wmma_f32_16x16x32_bf16(
                false, pa.v, false, vf.v, (short)0, acc[db], false, false);
        }
    }

    // Normalize + store f32 [b][h][t][d].
#pragma unroll
    for (int db = 0; db < 4; ++db)
#pragma unroll
        for (int r = 0; r < 8; ++r) {
            int m = r + ((lane & 16) ? 8 : 0);
            int qg = qw0 + m;
            int d = db * 16 + (lane & 15);
            ows[base + (size_t)qg * DK + d] = acc[db][r] / rl[r];
        }
}

// ---------------------------------------------------------------------------
// Kernel 4: out = attn_out @ Wo  (M=4096, N=768, K=768), gathering the
// [b][h][t][d] layout back into [b*t][h*64+d] rows while staging.
// ---------------------------------------------------------------------------
__global__ __launch_bounds__(128) void oproj_gemm_kernel(
    const float* __restrict__ aws, const float* __restrict__ wo,
    float* __restrict__ out)
{
    __shared__ unsigned short As[64][36];
    __shared__ unsigned short Bst[64][36];

    const int tid  = threadIdx.x;
    const int wave = tid >> 5;
    const int lane = tid & 31;
    const int wr = wave >> 1, wc = wave & 1;
    const int m0 = blockIdx.y * 64;
    const int n0 = blockIdx.x * 64;

    v8f c[2][2];
    c[0][0] = {}; c[0][1] = {}; c[1][0] = {}; c[1][1] = {};

    for (int kt = 0; kt < DM / 32; ++kt) {
        __syncthreads();
#pragma unroll
        for (int i = 0; i < 4; ++i) {
            int e = tid + i * 128;
            // A tile: row gm of [b*t][h*64+d] gathered from [b][h][t][d].
            int ar = e >> 3, ak = (e & 7) * 4;       // ak: 4 consecutive d
            int gm = m0 + ar;
            int bidx = gm >> 11, t = gm & 2047;
            int cidx = kt * 32 + ak;
            int h = cidx >> 6, d = cidx & 63;
            float4 av = *(const float4*)
                &aws[((size_t)(bidx * NH + h) * TT + t) * DK + d];
            uint2 ap;
            ap.x = pack_bf16x2(av.x, av.y);
            ap.y = pack_bf16x2(av.z, av.w);
            *(uint2*)&As[ar][ak] = ap;
            // B tile (transposed store).
            int br = e >> 4, bc = (e & 15) * 4;
            float4 bv = *(const float4*)&wo[(size_t)(kt * 32 + br) * DM + n0 + bc];
            Bst[bc + 0][br] = f32_to_bf16(bv.x);
            Bst[bc + 1][br] = f32_to_bf16(bv.y);
            Bst[bc + 2][br] = f32_to_bf16(bv.z);
            Bst[bc + 3][br] = f32_to_bf16(bv.w);
        }
        __syncthreads();

        Frag16 a[2], b[2];
#pragma unroll
        for (int mb = 0; mb < 2; ++mb)
#pragma unroll
            for (int vv = 0; vv < 8; ++vv) {
                int m = wr * 32 + mb * 16 + (lane & 15);
                a[mb].u[vv] = *(const unsigned*)&As[m][kofs(lane, vv)];
            }
#pragma unroll
        for (int nb = 0; nb < 2; ++nb)
#pragma unroll
            for (int vv = 0; vv < 8; ++vv) {
                int n = wc * 32 + nb * 16 + (lane & 15);
                b[nb].u[vv] = *(const unsigned*)&Bst[n][kofs(lane, vv)];
            }
#pragma unroll
        for (int mb = 0; mb < 2; ++mb)
#pragma unroll
            for (int nb = 0; nb < 2; ++nb)
                c[mb][nb] = __builtin_amdgcn_wmma_f32_16x16x32_bf16(
                    false, a[mb].v, false, b[nb].v, (short)0, c[mb][nb],
                    false, false);
    }

#pragma unroll
    for (int mb = 0; mb < 2; ++mb)
#pragma unroll
        for (int nb = 0; nb < 2; ++nb)
#pragma unroll
            for (int r = 0; r < 8; ++r) {
                int mg = m0 + wr * 32 + mb * 16 + r + ((lane & 16) ? 8 : 0);
                int ng = n0 + wc * 32 + nb * 16 + (lane & 15);
                out[(size_t)mg * DM + ng] = c[mb][nb][r];
            }
}

// ---------------------------------------------------------------------------
extern "C" void kernel_launch(void* const* d_in, const int* in_sizes, int n_in,
                              void* d_out, int out_size, void* d_ws, size_t ws_size,
                              hipStream_t stream)
{
    const float* x    = (const float*)d_in[0];   // [B*T, 768]
    const float* wqkv = (const float*)d_in[1];   // [768, 2304]
    const float* wo   = (const float*)d_in[2];   // [768, 768]
    float* out = (float*)d_out;                  // [B*T, 768]

    const size_t QK = (size_t)BB * NH * TT * DK; // 3,145,728 elems
    unsigned short* qb = (unsigned short*)d_ws;
    unsigned short* kb = qb + QK;
    unsigned short* vb = kb + QK;
    float* ows = (float*)(vb + QK);              // f32 attention output

    // 1) QKV projection + head scatter (bf16).
    qkv_gemm_kernel<<<dim3(QKVN / 64, (BB * TT) / 64), 128, 0, stream>>>(
        x, wqkv, qb, kb, vb);

    // 2) RoPE in place on Q and K.
    {
        int total = 2 * BB * NH * TT * 32;
        rope_kernel<<<(total + 255) / 256, 256, 0, stream>>>(qb, kb);
    }

    // 3) Causal flash attention (TDM K-tile staging when available).
    attn_kernel<<<dim3(TT / 128, BB * NH), 256, 0, stream>>>(qb, kb, vb, ows);

    // 4) Output projection.
    oproj_gemm_kernel<<<dim3(DM / 64, (BB * TT) / 64), 128, 0, stream>>>(
        ows, wo, out);
}
